// mLSTMblock_22582938043165
// MI455X (gfx1250) — compile-verified, matrix operand order checked by reading, not run on hardware
//
#include <hip/hip_runtime.h>
#include <math.h>

// Problem constants (reference: B=2, S=2048, D=256, H=512, K=25, DEPTH=4, HB=128)
#define Bsz 2
#define Sq  2048
#define Dd  256
#define Hh  512
#define Kk  25
#define HBb 128
#define RR  (Bsz*Sq)   // 4096 rows

typedef _Float16 v16h __attribute__((ext_vector_type(16)));
typedef _Float16 h8   __attribute__((ext_vector_type(8)));
typedef float    v8f  __attribute__((ext_vector_type(8)));
typedef int      v4i  __attribute__((ext_vector_type(4)));

union FragU { v16h v; h8 h[2]; };

#ifndef __has_builtin
#define __has_builtin(x) 0
#endif
#if __has_builtin(__builtin_amdgcn_global_load_async_to_lds_b128) && \
    __has_builtin(__builtin_amdgcn_s_wait_asynccnt)
#define HAVE_ASYNC 1
#else
#define HAVE_ASYNC 0
#endif

// 16-byte global->LDS copy. Async path uses GLOBAL_LOAD_ASYNC_TO_LDS_B128
// (ASYNCcnt-tracked, no VGPR round trip). Builtin signature (from clang
// diagnostic): (AS1 v4i* src, AS3 v4i* dst, imm offset, imm cpol).
// LDS generic VA: low 32 bits are the LDS byte offset (ISA 10.2), so an
// integer cast yields the AS3 pointer; global generic VA == 64-bit VA for AS1.
__device__ __forceinline__ void cp16(_Float16* dst, const _Float16* src) {
#if HAVE_ASYNC
  __builtin_amdgcn_global_load_async_to_lds_b128(
      (__attribute__((address_space(1))) v4i*)(unsigned long long)(const void*)src,
      (__attribute__((address_space(3))) v4i*)(unsigned int)(unsigned long long)(void*)dst,
      0, 0);
#else
  *(h8*)dst = *(const h8*)src;
#endif
}
__device__ __forceinline__ void stage_fence() {
#if HAVE_ASYNC
  __builtin_amdgcn_s_wait_asynccnt(0);
#endif
  __syncthreads();
}

// Build a 16x32 f16 WMMA fragment for this lane.
// Per CDNA5 ISA layout (16-bit A 16x32 / B 32x16):
//   lane half 0: elems 0..7 -> k = kb+0..7,   elems 8..15 -> k = kb+16..23
//   lane half 1: elems 0..7 -> k = kb+8..15,  elems 8..15 -> k = kb+24..31
__device__ __forceinline__ v16h load_frag(const _Float16* rowp, int kb, int half) {
  FragU f;
  f.h[0] = *(const h8*)(rowp + kb + half*8);
  f.h[1] = *(const h8*)(rowp + kb + 16 + half*8);
  return f.v;
}

// ---------------- block reductions (power-of-2 blockDim) ----------------
__device__ __forceinline__ float blk_sum(float v, float* sm) {
  int t = threadIdx.x;
  sm[t] = v; __syncthreads();
  for (int s = blockDim.x >> 1; s > 0; s >>= 1) {
    if (t < s) sm[t] += sm[t + s];
    __syncthreads();
  }
  float r = sm[0]; __syncthreads();
  return r;
}
__device__ __forceinline__ float blk_max(float v, float* sm) {
  int t = threadIdx.x;
  sm[t] = v; __syncthreads();
  for (int s = blockDim.x >> 1; s > 0; s >>= 1) {
    if (t < s) sm[t] = fmaxf(sm[t], sm[t + s]);
    __syncthreads();
  }
  float r = sm[0]; __syncthreads();
  return r;
}

// ---------------- weight prep (f32 -> f16, transposed layouts) ----------------
__global__ void transpose_to_f16(const float* __restrict__ W, _Float16* __restrict__ Wt,
                                 int Kd, int N) {
  int idx = blockIdx.x * blockDim.x + threadIdx.x;
  if (idx >= Kd * N) return;
  int k = idx / N, n = idx % N;
  Wt[(size_t)n * Kd + k] = (_Float16)W[idx];     // Wt[n][k]
}

// Wc [O][I][K] f32 -> Wck [k][o][i] f16 (per-tap B matrices, lane-contiguous in i)
__global__ void convw_to_f16(const float* __restrict__ Wc, _Float16* __restrict__ Wck) {
  int idx = blockIdx.x * blockDim.x + threadIdx.x;
  if (idx >= Hh * Hh * Kk) return;
  int k = idx % Kk;
  int i = (idx / Kk) % Hh;
  int o = idx / (Kk * Hh);
  Wck[(size_t)k * Hh * Hh + (size_t)o * Hh + i] = (_Float16)Wc[idx];
}

// Wb [d][i][o] f32 -> Wt [(d*HB+o)][i] f16 (block-diagonal, transposed)
__global__ void bdw_to_f16(const float* __restrict__ Wb, _Float16* __restrict__ Wt) {
  int idx = blockIdx.x * blockDim.x + threadIdx.x;
  if (idx >= 4 * HBb * HBb) return;
  int o = idx % HBb;
  int i = (idx / HBb) % HBb;
  int d = idx / (HBb * HBb);
  Wt[((size_t)(d * HBb + o)) * HBb + i] = (_Float16)Wb[idx];
}

// ---------------- row LayerNorm over D=256 ----------------
__global__ void ln256_k(const float* __restrict__ X, const float* __restrict__ g,
                        const float* __restrict__ b, float* oF, _Float16* oH) {
  __shared__ float sm[256];
  int c = threadIdx.x;
  size_t base = (size_t)blockIdx.x * Dd;
  float x = X[base + c];
  const float invD = 1.f / Dd;
  float m  = blk_sum(x, sm) * invD;
  float vv = blk_sum((x - m) * (x - m), sm) * invD;
  float rs = rsqrtf(vv + 1e-5f);
  float y = (x - m) * rs * g[c] + b[c];
  if (oF) oF[base + c] = y;
  if (oH) oH[base + c] = (_Float16)y;
}

// ------- generic WMMA GEMM, 2x2 register-blocked: C[M,N] = A@Bt^T + bias -----
// grid.x = M/32, grid.y = N/128; block = 128 (4 waves).
// Each wave computes a 32x32 macro-tile (2 m-tiles x 2 n-tiles), so each A and
// B fragment feeds two wmma (2 loads/wmma instead of 4).
// bdHB != 0 => block-diagonal: A column window slides with output-column block.
__global__ void gemm_wmma_k(const _Float16* __restrict__ A, int lda, int bdHB,
                            const _Float16* __restrict__ Bt,
                            const float* __restrict__ bias,
                            float* Cf, _Float16* Ch,
                            int ldc, int Kd, int act) {
  __shared__ __align__(16) _Float16 As[32 * (512 + 8)];
  const int tid  = threadIdx.x;
  const int row0 = blockIdx.x * 32;
  const int n0   = blockIdx.y * 128;
  const int aOff = bdHB ? (n0 / bdHB) * bdHB : 0;
  const int stride = Kd + 8;
  const int chunks = Kd >> 3;                 // 16-byte chunks per row
  for (int idx = tid; idx < 32 * chunks; idx += blockDim.x) {
    int r = idx / chunks, c = idx % chunks;
    cp16(As + r * stride + c * 8,
         A + (size_t)(row0 + r) * lda + aOff + c * 8);
  }
  stage_fence();

  const int wave = tid >> 5, lane = tid & 31;
  const int half = lane >> 4, l15 = lane & 15;
  const int col0 = n0 + wave * 32 + l15;      // n-tile 0
  const int col1 = col0 + 16;                 // n-tile 1
  float b0 = bias ? bias[col0] : 0.f;
  float b1 = bias ? bias[col1] : 0.f;
  v8f acc[2][2];
  #pragma unroll
  for (int i = 0; i < 8; ++i) {
    acc[0][0][i] = b0; acc[1][0][i] = b0;
    acc[0][1][i] = b1; acc[1][1][i] = b1;
  }

  const _Float16* arow0 = As + l15 * stride;          // m-tile 0 (LDS)
  const _Float16* arow1 = As + (l15 + 16) * stride;   // m-tile 1 (LDS)
  const _Float16* brow0 = Bt + (size_t)col0 * Kd;     // n-tile 0 (global)
  const _Float16* brow1 = Bt + (size_t)col1 * Kd;     // n-tile 1 (global)
  for (int kb = 0; kb < Kd; kb += 32) {
    v16h a0 = load_frag(arow0, kb, half);
    v16h a1 = load_frag(arow1, kb, half);
    v16h bf0 = load_frag(brow0, kb, half);
    v16h bf1 = load_frag(brow1, kb, half);
    acc[0][0] = __builtin_amdgcn_wmma_f32_16x16x32_f16(false, a0, false, bf0,
                                                       (short)0, acc[0][0], false, false);
    acc[0][1] = __builtin_amdgcn_wmma_f32_16x16x32_f16(false, a0, false, bf1,
                                                       (short)0, acc[0][1], false, false);
    acc[1][0] = __builtin_amdgcn_wmma_f32_16x16x32_f16(false, a1, false, bf0,
                                                       (short)0, acc[1][0], false, false);
    acc[1][1] = __builtin_amdgcn_wmma_f32_16x16x32_f16(false, a1, false, bf1,
                                                       (short)0, acc[1][1], false, false);
  }
  #pragma unroll
  for (int mt = 0; mt < 2; ++mt)
    #pragma unroll
    for (int nt = 0; nt < 2; ++nt)
      #pragma unroll
      for (int r = 0; r < 8; ++r) {
        int m = mt * 16 + r + 8 * half;               // C layout: VGPR r -> M=r(+8)
        int c = (nt == 0) ? col0 : col1;
        float v = acc[mt][nt][r];
        if (act == 1) v = v / (1.f + __expf(-v));     // SiLU
        size_t o = (size_t)(row0 + m) * ldc + c;
        if (Cf) Cf[o] = v;
        if (Ch) Ch[o] = (_Float16)v;
      }
}

// ---------------- causal conv as 25 shifted WMMA GEMMs, 2x2 blocked ----------
// cv[t,o] = sum_{k=0..24} sum_i left[t+k-24, i] * Wc[o,i,k]; left_left = silu(cv)
// 56-row f16 LDS tile (32 output rows + 24 causal history) reused by all taps.
__global__ void conv_wmma_k(const _Float16* __restrict__ L16,
                            const _Float16* __restrict__ Wck,
                            _Float16* __restrict__ LL16) {
  __shared__ __align__(16) _Float16 Ls[56 * (512 + 8)];
  const int tid  = threadIdx.x;
  const int row0 = blockIdx.x * 32;
  const int b    = row0 / Sq;
  const int t0   = row0 % Sq;

  for (int idx = tid; idx < 56 * 64; idx += blockDim.x) { // 64 chunks of 8 halves/row
    int j = idx / 64, c = idx % 64;
    int t = t0 - (Kk - 1) + j;
    _Float16* dst = Ls + j * (512 + 8) + c * 8;
    if (t >= 0) {
      cp16(dst, L16 + ((size_t)(b * Sq + t)) * Hh + c * 8);
    } else {
      h8 z;
      #pragma unroll
      for (int e = 0; e < 8; ++e) z[e] = (_Float16)0.f;  // causal zero pad
      *(h8*)dst = z;
    }
  }
  stage_fence();

  const int wave = tid >> 5, lane = tid & 31;
  const int half = lane >> 4, l15 = lane & 15;
  const int col0 = blockIdx.y * 128 + wave * 32 + l15;
  const int col1 = col0 + 16;
  v8f acc[2][2];
  #pragma unroll
  for (int i = 0; i < 8; ++i) {
    acc[0][0][i] = 0.f; acc[0][1][i] = 0.f;
    acc[1][0][i] = 0.f; acc[1][1][i] = 0.f;
  }

  for (int k = 0; k < Kk; ++k) {
    const _Float16* arow0 = Ls + (size_t)(l15 + k) * (512 + 8);
    const _Float16* arow1 = Ls + (size_t)(l15 + 16 + k) * (512 + 8);
    const _Float16* brow0 = Wck + (size_t)k * Hh * Hh + (size_t)col0 * Hh;
    const _Float16* brow1 = Wck + (size_t)k * Hh * Hh + (size_t)col1 * Hh;
    #pragma unroll 4
    for (int kb = 0; kb < Hh; kb += 32) {
      v16h a0 = load_frag(arow0, kb, half);
      v16h a1 = load_frag(arow1, kb, half);
      v16h bf0 = load_frag(brow0, kb, half);
      v16h bf1 = load_frag(brow1, kb, half);
      acc[0][0] = __builtin_amdgcn_wmma_f32_16x16x32_f16(false, a0, false, bf0,
                                                         (short)0, acc[0][0], false, false);
      acc[0][1] = __builtin_amdgcn_wmma_f32_16x16x32_f16(false, a0, false, bf1,
                                                         (short)0, acc[0][1], false, false);
      acc[1][0] = __builtin_amdgcn_wmma_f32_16x16x32_f16(false, a1, false, bf0,
                                                         (short)0, acc[1][0], false, false);
      acc[1][1] = __builtin_amdgcn_wmma_f32_16x16x32_f16(false, a1, false, bf1,
                                                         (short)0, acc[1][1], false, false);
    }
  }
  #pragma unroll
  for (int mt = 0; mt < 2; ++mt)
    #pragma unroll
    for (int nt = 0; nt < 2; ++nt)
      #pragma unroll
      for (int r = 0; r < 8; ++r) {
        int m = mt * 16 + r + 8 * half;
        int c = (nt == 0) ? col0 : col1;
        float v = acc[mt][nt][r];
        v = v / (1.f + __expf(-v));                    // SiLU
        LL16[(size_t)(row0 + m) * Hh + c] = (_Float16)v;
      }
}

// ---------------- gates + per-position ln(ct), ln(nt) ----------------
// Writes o_ over Yi, ln(ct) over kk, ln(nt) over v (each element read before write).
__global__ void gates_k(const float* Yi, const float* Yf, const float* Yo,
                        const float* vb, const float* kb,
                        const float* gi, const float* bgi,
                        const float* gf, const float* bgf,
                        const float* go, const float* bgo,
                        const float* gc, const float* bgc,
                        const float* gn, const float* bgn,
                        const float* ct1, const float* nt1,
                        float* o_out, float* lnct, float* lnnt) {
  __shared__ float sm[512];
  int h = threadIdx.x;
  size_t base = (size_t)blockIdx.x * Hh;
  const float invH = 1.f / Hh;
  float yi = Yi[base + h], yf = Yf[base + h], yo = Yo[base + h];
  float vv = vb[base + h], kk = kb[base + h];

  float m  = blk_sum(yi, sm) * invH;
  float va = blk_sum((yi - m) * (yi - m), sm) * invH;
  float i_ = __expf((yi - m) * rsqrtf(va + 1e-5f) * gi[h] + bgi[h]);

  m  = blk_sum(yf, sm) * invH;
  va = blk_sum((yf - m) * (yf - m), sm) * invH;
  float f_ = __expf((yf - m) * rsqrtf(va + 1e-5f) * gf[h] + bgf[h]);

  m  = blk_sum(yo, sm) * invH;
  va = blk_sum((yo - m) * (yo - m), sm) * invH;
  float olog = (yo - m) * rsqrtf(va + 1e-5f) * go[h] + bgo[h];
  float o_ = 1.f / (1.f + __expf(-olog));

  float cte = f_ * ct1[h] + i_ * vv * kk;
  m  = blk_sum(cte, sm) * invH;
  va = blk_sum((cte - m) * (cte - m), sm) * invH;
  float lc = (cte - m) * rsqrtf(va + 1e-5f) * gc[h] + bgc[h];

  float nte = f_ * nt1[h] + i_ * kk;
  m  = blk_sum(nte, sm) * invH;
  va = blk_sum((nte - m) * (nte - m), sm) * invH;
  float ln_ = (nte - m) * rsqrtf(va + 1e-5f) * gn[h] + bgn[h];

  o_out[base + h] = o_;
  lnct[base + h]  = lc;
  lnnt[base + h]  = ln_;
}

// Deterministic column means over 4096 rows -> ct_mean, nt_mean [H]
__global__ void colmean_k(const float* lnct, const float* lnnt, float* ctm, float* ntm) {
  __shared__ float sm[256];
  int h = blockIdx.x, t = threadIdx.x;
  float s1 = 0.f, s2 = 0.f;
  for (int r = t; r < RR; r += 256) {
    s1 += lnct[(size_t)r * Hh + h];
    s2 += lnnt[(size_t)r * Hh + h];
  }
  float S1 = blk_sum(s1, sm);
  float S2 = blk_sum(s2, sm);
  if (t == 0) { ctm[h] = S1 / RR; ntm[h] = S2 / RR; }
}

// Deterministic global max of nt_mean[h] * q[r,h]
__global__ void max_partial_k(const float* q, const float* ntm, float* part) {
  __shared__ float sm[256];
  size_t i = (size_t)blockIdx.x * blockDim.x + threadIdx.x;
  size_t step = (size_t)gridDim.x * blockDim.x;
  float mx = -3.4e38f;
  for (; i < (size_t)RR * Hh; i += step)
    mx = fmaxf(mx, ntm[i % Hh] * q[i]);
  mx = blk_max(mx, sm);
  if (threadIdx.x == 0) part[blockIdx.x] = mx;
}
__global__ void max_final_k(const float* part, float* mv) {
  __shared__ float sm[256];
  float mx = blk_max(part[threadIdx.x], sm);
  if (threadIdx.x == 0) mv[0] = mx;
}

// ht -> LN(ggn) -> *right -> LN(gout) -> u (f16 for final GEMM)
__global__ void outpre_k(const float* o_, const float* q, const float* lskip,
                         const float* right, const float* ctm, const float* mv,
                         const float* ggn, const float* bggn,
                         const float* gout, const float* bgout,
                         _Float16* u16) {
  __shared__ float sm[512];
  int h = threadIdx.x;
  size_t base = (size_t)blockIdx.x * Hh;
  const float invH = 1.f / Hh;
  float ht = o_[base + h] * (ctm[h] * q[base + h] / mv[0]);
  float a  = ht + lskip[base + h];
  float m  = blk_sum(a, sm) * invH;
  float va = blk_sum((a - m) * (a - m), sm) * invH;
  float l2 = (a - m) * rsqrtf(va + 1e-5f) * ggn[h] + bggn[h];
  float t2 = l2 * right[base + h];
  m  = blk_sum(t2, sm) * invH;
  va = blk_sum((t2 - m) * (t2 - m), sm) * invH;
  float u = (t2 - m) * rsqrtf(va + 1e-5f) * gout[h] + bgout[h];
  u16[base + h] = (_Float16)u;
}

// ---------------- launch ----------------
extern "C" void kernel_launch(void* const* d_in, const int* in_sizes, int n_in,
                              void* d_out, int out_size, void* d_ws, size_t ws_size,
                              hipStream_t stream) {
  (void)in_sizes; (void)n_in; (void)out_size; (void)ws_size;
  const float* x    = (const float*)d_in[0];
  const float* ln_g = (const float*)d_in[1];
  const float* ln_b = (const float*)d_in[2];
  const float* Wl = (const float*)d_in[3];   const float* bl = (const float*)d_in[4];
  const float* Wr = (const float*)d_in[5];   const float* br = (const float*)d_in[6];
  const float* Wc = (const float*)d_in[7];   /* bc = d_in[8] unused by reference */
  const float* Wskip=(const float*)d_in[9];  const float* bskip=(const float*)d_in[10];
  const float* Wq = (const float*)d_in[11];  const float* bq = (const float*)d_in[12];
  const float* Wk = (const float*)d_in[13];  const float* bk = (const float*)d_in[14];
  const float* Wv = (const float*)d_in[15];  const float* bv = (const float*)d_in[16];
  const float* Wi = (const float*)d_in[17];  const float* bi = (const float*)d_in[18];
  const float* Wf = (const float*)d_in[19];  const float* bf = (const float*)d_in[20];
  const float* Wo = (const float*)d_in[21];  const float* bo = (const float*)d_in[22];
  const float* gi = (const float*)d_in[23];  const float* bgi= (const float*)d_in[24];
  const float* gf = (const float*)d_in[25];  const float* bgf= (const float*)d_in[26];
  const float* go = (const float*)d_in[27];  const float* bgo= (const float*)d_in[28];
  const float* gc = (const float*)d_in[29];  const float* bgc= (const float*)d_in[30];
  const float* gn = (const float*)d_in[31];  const float* bgn= (const float*)d_in[32];
  const float* ggn= (const float*)d_in[33];  const float* bggn=(const float*)d_in[34];
  const float* gout=(const float*)d_in[35];  const float* bgout=(const float*)d_in[36];
  const float* Wp = (const float*)d_in[37];  const float* bp = (const float*)d_in[38];
  const float* gp = (const float*)d_in[39];  const float* bgp= (const float*)d_in[40];
  const float* ct1= (const float*)d_in[41];  const float* nt1= (const float*)d_in[42];

  char* ws = (char*)d_ws;
  size_t off = 0;
  auto alloc = [&](size_t bytes) -> char* {
    off = (off + 255) & ~(size_t)255;
    char* p = ws + off;
    off += bytes;
    return p;
  };

  _Float16* xn16 = (_Float16*)alloc((size_t)RR * Dd * 2);
  _Float16* Wl_t = (_Float16*)alloc((size_t)Hh * Dd * 2);
  _Float16* Wr_t = (_Float16*)alloc((size_t)Hh * Dd * 2);
  _Float16* Wck  = (_Float16*)alloc((size_t)Kk * Hh * Hh * 2);
  _Float16* Wsk_t= (_Float16*)alloc((size_t)Hh * Hh * 2);
  _Float16* Wi_t = (_Float16*)alloc((size_t)Hh * Hh * 2);
  _Float16* Wf_t = (_Float16*)alloc((size_t)Hh * Hh * 2);
  _Float16* Wo_t = (_Float16*)alloc((size_t)Hh * Hh * 2);
  _Float16* Wq_t = (_Float16*)alloc((size_t)Hh * HBb * 2);
  _Float16* Wk_t = (_Float16*)alloc((size_t)Hh * HBb * 2);
  _Float16* Wv_t = (_Float16*)alloc((size_t)Hh * HBb * 2);
  _Float16* Wp_t = (_Float16*)alloc((size_t)Dd * Hh * 2);
  _Float16* L16  = (_Float16*)alloc((size_t)RR * Hh * 2);  // left f16; later reused as u16
  _Float16* LL16 = (_Float16*)alloc((size_t)RR * Hh * 2);  // left_left f16; later reused as Yp f32
  float* right32 = (float*)alloc((size_t)RR * Hh * 4);
  float* lskip32 = (float*)alloc((size_t)RR * Hh * 4);
  float* Yi32    = (float*)alloc((size_t)RR * Hh * 4);     // becomes o_
  float* Yf32    = (float*)alloc((size_t)RR * Hh * 4);
  float* Yo32    = (float*)alloc((size_t)RR * Hh * 4);
  float* q32     = (float*)alloc((size_t)RR * Hh * 4);
  float* kk32    = (float*)alloc((size_t)RR * Hh * 4);     // becomes ln(ct)
  float* v32     = (float*)alloc((size_t)RR * Hh * 4);     // becomes ln(nt)
  float* ctm     = (float*)alloc(Hh * 4);
  float* ntm     = (float*)alloc(Hh * 4);
  float* part    = (float*)alloc(256 * 4);
  float* mv      = (float*)alloc(4);

  const int T = 256;
  // Weight prep
  transpose_to_f16<<<(Dd*Hh + T-1)/T, T, 0, stream>>>(Wl, Wl_t, Dd, Hh);
  transpose_to_f16<<<(Dd*Hh + T-1)/T, T, 0, stream>>>(Wr, Wr_t, Dd, Hh);
  transpose_to_f16<<<(Hh*Hh + T-1)/T, T, 0, stream>>>(Wskip, Wsk_t, Hh, Hh);
  transpose_to_f16<<<(Hh*Hh + T-1)/T, T, 0, stream>>>(Wi, Wi_t, Hh, Hh);
  transpose_to_f16<<<(Hh*Hh + T-1)/T, T, 0, stream>>>(Wf, Wf_t, Hh, Hh);
  transpose_to_f16<<<(Hh*Hh + T-1)/T, T, 0, stream>>>(Wo, Wo_t, Hh, Hh);
  transpose_to_f16<<<(Hh*Dd + T-1)/T, T, 0, stream>>>(Wp, Wp_t, Hh, Dd);
  convw_to_f16<<<(Hh*Hh*Kk + T-1)/T, T, 0, stream>>>(Wc, Wck);
  bdw_to_f16<<<(4*HBb*HBb + T-1)/T, T, 0, stream>>>(Wq, Wq_t);
  bdw_to_f16<<<(4*HBb*HBb + T-1)/T, T, 0, stream>>>(Wk, Wk_t);
  bdw_to_f16<<<(4*HBb*HBb + T-1)/T, T, 0, stream>>>(Wv, Wv_t);

  // x -> LN -> f16
  ln256_k<<<RR, 256, 0, stream>>>(x, ln_g, ln_b, nullptr, xn16);

  dim3 gH(RR/32, Hh/128);                       // (128, 4)
  // left (f16 only), right = silu(...) (f32)
  gemm_wmma_k<<<gH, 128, 0, stream>>>(xn16, Dd, 0, Wl_t, bl, nullptr, L16, Hh, Dd, 0);
  gemm_wmma_k<<<gH, 128, 0, stream>>>(xn16, Dd, 0, Wr_t, br, right32, nullptr, Hh, Dd, 1);

  // causal conv + silu -> left_left f16
  conv_wmma_k<<<gH, 128, 0, stream>>>(L16, Wck, LL16);

  // dense H x H GEMMs from left_left
  gemm_wmma_k<<<gH, 128, 0, stream>>>(LL16, Hh, 0, Wsk_t, bskip, lskip32, nullptr, Hh, Hh, 0);
  gemm_wmma_k<<<gH, 128, 0, stream>>>(LL16, Hh, 0, Wi_t, bi, Yi32, nullptr, Hh, Hh, 0);
  gemm_wmma_k<<<gH, 128, 0, stream>>>(LL16, Hh, 0, Wf_t, bf, Yf32, nullptr, Hh, Hh, 0);
  gemm_wmma_k<<<gH, 128, 0, stream>>>(LL16, Hh, 0, Wo_t, bo, Yo32, nullptr, Hh, Hh, 0);
  // block-diagonal GEMMs (K=128, A window slides with the 128-col block)
  gemm_wmma_k<<<gH, 128, 0, stream>>>(LL16, Hh, HBb, Wq_t, bq, q32, nullptr, Hh, HBb, 0);
  gemm_wmma_k<<<gH, 128, 0, stream>>>(LL16, Hh, HBb, Wk_t, bk, kk32, nullptr, Hh, HBb, 0);
  gemm_wmma_k<<<gH, 128, 0, stream>>>(L16,  Hh, HBb, Wv_t, bv, v32, nullptr, Hh, HBb, 0);

  // gates + per-position ln(ct)/ln(nt) (in-place over Yi/kk/v)
  gates_k<<<RR, 512, 0, stream>>>(Yi32, Yf32, Yo32, v32, kk32,
                                  gi, bgi, gf, bgf, go, bgo, gc, bgc, gn, bgn,
                                  ct1, nt1, Yi32, kk32, v32);
  colmean_k<<<Hh, 256, 0, stream>>>(kk32, v32, ctm, ntm);
  max_partial_k<<<256, 256, 0, stream>>>(q32, ntm, part);
  max_final_k<<<1, 256, 0, stream>>>(part, mv);

  // ht, LN(ggn), *right, LN(gout) -> u f16 (reuse L16)
  outpre_k<<<RR, 512, 0, stream>>>(Yi32, q32, lskip32, right32, ctm, mv,
                                   ggn, bggn, gout, bgout, L16);

  // final projection u @ Wp + bp -> Yp (reuse LL16 space), then LN -> d_out
  float* Yp = (float*)LL16;
  gemm_wmma_k<<<dim3(RR/32, Dd/128), 128, 0, stream>>>(L16, Hh, 0, Wp_t, bp,
                                                       Yp, nullptr, Dd, Hh, 0);
  ln256_k<<<RR, 256, 0, stream>>>(Yp, gp, bgp, (float*)d_out, nullptr);
}
